// PaperGRUHead_59072980189901
// MI455X (gfx1250) — matrix operand order repaired and checked by reading, not compiled, not run on hardware
//
#include <hip/hip_runtime.h>

#define USE_TDM 1

// Problem dims
static constexpr int B      = 8192;
static constexpr int H      = 256;
static constexpr int G      = 768;   // 3*H gates
static constexpr int CTX    = 512;
static constexpr int IN_DIM = 514;
static constexpr int PRED   = 12;
static constexpr int T_OBS  = 8;

// GEMM tiling
static constexpr int KC   = 256;       // K chunk staged in LDS
static constexpr int SROW = KC + 8;    // padded LDS row stride (halves): 528B = 4 banks offset

typedef __attribute__((ext_vector_type(16))) _Float16 v16h;
typedef __attribute__((ext_vector_type(8)))  _Float16 v8h;
typedef __attribute__((ext_vector_type(8)))  float    v8f;
typedef __attribute__((ext_vector_type(4)))  int      v4i;
typedef __attribute__((ext_vector_type(8)))  int      v8i;

// ---------- conversions ----------
__global__ void cvt_f32_f16(const float* __restrict__ src, _Float16* __restrict__ dst, int n) {
  int i = blockIdx.x * 256 + threadIdx.x;
  if (i < n) dst[i] = (_Float16)src[i];
}

// Wih0 is [768, 514] row-major; extract first 512 columns into f16 [768, 512]
__global__ void cvt_wih0_ctx(const float* __restrict__ src, _Float16* __restrict__ dst) {
  int i = blockIdx.x * 256 + threadIdx.x;           // 768*512 threads
  int r = i >> 9, c = i & 511;
  dst[i] = (_Float16)src[r * IN_DIM + c];
}

// zero hidden states (fp32 + f16 mirrors), copy obs_traj[-1] into cur_pos
__global__ void init_state(float* __restrict__ h, _Float16* __restrict__ h16,
                           const float* __restrict__ obs_last, float* __restrict__ cur) {
  int i = blockIdx.x * 256 + threadIdx.x;
  if (i < 3 * B * H) { h[i] = 0.0f; h16[i] = (_Float16)0.0f; }
  if (i < B * 2) cur[i] = obs_last[i];
}

#if USE_TDM
// TDM: stage a [rows x tile_k] f16 tile (row stride = stride_elems in global) into LDS,
// padded by 4 DWORDs every 128 DWORDs so the LDS row stride becomes SROW halves.
// D# layout per CDNA5 ISA sec 8.3/8.4 (group0 128b, group1 256b; groups 2/3 zero => 2D).
__device__ __forceinline__ void tdm_load_weights(const _Float16* gsrc, unsigned lds_byte,
                                                 int stride_elems, int tile_k, int rows) {
  unsigned long long ga = (unsigned long long)(uintptr_t)gsrc;
  v4i g0;
  g0[0] = 1;                                        // count=1, user-mode, no gather
  g0[1] = (int)lds_byte;                            // lds_addr (bytes)
  g0[2] = (int)(unsigned)(ga & 0xffffffffull);      // global_addr[31:0]
  g0[3] = (int)((unsigned)((ga >> 32) & 0x01ffffffull) | 0x80000000u); // addr[56:32] | type=2
  v8i g1;
  g1[0] = (1 << 16)      // data_size = 1 -> 2 bytes
        | (1 << 20)      // pad_enable
        | (6 << 22)      // pad_interval: code 6 = every 128 DWORDs (512B row)
        | (3 << 25);     // pad_amount:   code 3 = 4 DWORDs (8 halves)
  g1[1] = (stride_elems & 0xffff) << 16;                         // tensor_dim0[15:0]
  g1[2] = ((stride_elems >> 16) & 0xffff) | ((rows & 0xffff) << 16); // dim0 hi | tensor_dim1 lo
  g1[3] = (tile_k & 0xffff) << 16;                               // tensor_dim1 hi | tile_dim0
  g1[4] = rows & 0xffff;                                         // tile_dim1 | tile_dim2=0
  g1[5] = stride_elems;                                          // tensor_dim0_stride[31:0]
  g1[6] = 0;
  g1[7] = 0;
  v4i gz = {0, 0, 0, 0};
  asm volatile("tensor_load_to_lds %0, %1, %2, %3"
               :: "s"(g0), "s"(g1), "s"(gz), "s"(gz) : "memory");
}
#endif

// ---------- WMMA GEMM:  C[b,g] = sum_k X[b,k]*W[g,k] (+bias[g]) ----------
// X: [B,K] f16 row-major;  W: [G,K] f16 row-major;  C: [B,G] f32
// Block = 256 threads (8 wave32). Block tile M=128, N=64. Wave tile 64x16.
// Weight tile staged in LDS (TDM), A streamed from global.
template <int K>
__global__ __launch_bounds__(256) void gemm_wmma(const _Float16* __restrict__ X,
                                                 const _Float16* __restrict__ W,
                                                 const float* __restrict__ bias,
                                                 float* __restrict__ C) {
  __shared__ _Float16 sW[64 * SROW];                // 33 KB

  const int lane  = threadIdx.x & 31;
  const int wid   = threadIdx.x >> 5;
  const int nsub  = wid & 3;           // 4 waves across N
  const int msub  = wid >> 2;          // 2 waves across M
  const int khi   = lane >> 4;         // lane half (0/1)
  const int mlo   = lane & 15;
  const int gl    = nsub * 16 + mlo;                       // local output column
  const int gblk  = blockIdx.x * 64;
  const int g     = gblk + gl;                             // global output column
  const int mbase = blockIdx.y * 128 + msub * 64;

  v8f acc[4] = {};

  for (int kc = 0; kc < K; kc += KC) {
    // ---- stage 64 x KC weight chunk into LDS ----
#if USE_TDM
    if (wid == 0) {
      tdm_load_weights(W + (size_t)gblk * K + kc, (unsigned)(uintptr_t)&sW[0],
                       K, KC, 64);
      __builtin_amdgcn_s_wait_tensorcnt(0);
    }
    __syncthreads();
#else
    {
      int r = threadIdx.x >> 2;            // 0..63 row
      int seg = (threadIdx.x & 3) * 64;    // 64-half segment
      const _Float16* src = W + (size_t)(gblk + r) * K + kc + seg;
      _Float16* dst = &sW[r * SROW + seg];
#pragma unroll
      for (int u = 0; u < 64; u += 8) *(v8h*)(dst + u) = *(const v8h*)(src + u);
    }
    __syncthreads();
#endif

    // ---- MAC over the chunk ----
    for (int k0 = 0; k0 < KC; k0 += 32) {
      // B fragment (32x16 KxN): lane n=gl, elem e -> k = k0 + 16*khi + e (from LDS)
      v16h bfrag = *(const v16h*)(&sW[gl * SROW + k0 + khi * 16]);

      // A fragment loader (16x32 MxK): lane row m = mlo; elem e -> k0+16*khi+e
      auto lda = [&](int i) -> v16h {
        const _Float16* xp = X + (size_t)(mbase + i * 16 + mlo) * K + kc + k0 + khi * 16;
        v8h lo = *(const v8h*)xp;
        v8h hi = *(const v8h*)(xp + 8);
        v16h a;
#pragma unroll
        for (int e = 0; e < 8; ++e) { a[e] = lo[e]; a[8 + e] = hi[e]; }
        return a;
      };
      // Manually unrolled so reuse flags are integer constants (Sema requirement).
      // reuse_b hints that the NEXT wmma reuses the same B operand (true for the
      // first three of each group, false on the last which precedes a B switch).
      acc[0] = __builtin_amdgcn_wmma_f32_16x16x32_f16(
          false, lda(0), false, bfrag, (short)0, acc[0], false, true);
      acc[1] = __builtin_amdgcn_wmma_f32_16x16x32_f16(
          false, lda(1), false, bfrag, (short)0, acc[1], false, true);
      acc[2] = __builtin_amdgcn_wmma_f32_16x16x32_f16(
          false, lda(2), false, bfrag, (short)0, acc[2], false, true);
      acc[3] = __builtin_amdgcn_wmma_f32_16x16x32_f16(
          false, lda(3), false, bfrag, (short)0, acc[3], false, false);
    }
    __syncthreads();
  }

  const float bv = bias ? bias[g] : 0.0f;
#pragma unroll
  for (int i = 0; i < 4; ++i) {
    // C/D layout: lane L, vgpr v -> row m = v + 8*khi, col n = L&15
    int m0 = mbase + i * 16 + khi * 8;
#pragma unroll
    for (int v = 0; v < 8; ++v)
      C[(size_t)(m0 + v) * G + g] = acc[i][v] + bv;
  }
}

// ---------- fused GRU gate elementwise ----------
// gi: [B,768] raw X@Wih^T (layer0: gi0_base already includes bih0)
// gh: [B,768] raw h@Whh^T
// Wih0_full/cur non-null only for layer 0 (adds cur_pos rank-2 contribution).
__global__ void gru_ew(const float* __restrict__ gi, const float* __restrict__ bih,
                       const float* __restrict__ gh, const float* __restrict__ bhh,
                       const float* __restrict__ Wih0_full, const float* __restrict__ cur,
                       float* __restrict__ h, _Float16* __restrict__ h16) {
  int idx = blockIdx.x * 256 + threadIdx.x;   // B*H threads
  int b = idx >> 8, j = idx & 255;
  const float* gib = gi + (size_t)b * G;
  const float* ghb = gh + (size_t)b * G;
  float ir = gib[j], iz = gib[j + 256], inn = gib[j + 512];
  if (bih) { ir += bih[j]; iz += bih[j + 256]; inn += bih[j + 512]; }
  if (Wih0_full) {
    float p0 = cur[b * 2], p1 = cur[b * 2 + 1];
    ir  += p0 * Wih0_full[(size_t)j * IN_DIM + 512]         + p1 * Wih0_full[(size_t)j * IN_DIM + 513];
    iz  += p0 * Wih0_full[(size_t)(j + 256) * IN_DIM + 512] + p1 * Wih0_full[(size_t)(j + 256) * IN_DIM + 513];
    inn += p0 * Wih0_full[(size_t)(j + 512) * IN_DIM + 512] + p1 * Wih0_full[(size_t)(j + 512) * IN_DIM + 513];
  }
  float hr = ghb[j] + bhh[j];
  float hz = ghb[j + 256] + bhh[j + 256];
  float hn = ghb[j + 512] + bhh[j + 512];
  float r = 1.0f / (1.0f + expf(-(ir + hr)));
  float z = 1.0f / (1.0f + expf(-(iz + hz)));
  float n = tanhf(inn + r * hn);
  float hv = (1.0f - z) * n + z * h[idx];
  h[idx]   = hv;
  h16[idx] = (_Float16)hv;
}

// ---------- final projection: out = h2 @ fc_w^T + fc_b ; also next cur_pos ----------
__global__ void fc_out(const float* __restrict__ h, const float* __restrict__ fc_w,
                       const float* __restrict__ fc_b, float* __restrict__ out_t,
                       float* __restrict__ cur) {
  int b = blockIdx.x * 256 + threadIdx.x;
  if (b >= B) return;
  const float* hb = h + (size_t)b * H;
  float a0 = 0.0f, a1 = 0.0f;
#pragma unroll 4
  for (int k = 0; k < H; ++k) {
    float hv = hb[k];
    a0 += hv * fc_w[k];
    a1 += hv * fc_w[H + k];
  }
  a0 += fc_b[0]; a1 += fc_b[1];
  out_t[b * 2] = a0; out_t[b * 2 + 1] = a1;
  cur[b * 2] = a0;   cur[b * 2 + 1] = a1;
}

extern "C" void kernel_launch(void* const* d_in, const int* in_sizes, int n_in,
                              void* d_out, int out_size, void* d_ws, size_t ws_size,
                              hipStream_t stream) {
  const float* raw_ctx = (const float*)d_in[0];
  const float* obs     = (const float*)d_in[1];
  const float* Wih0 = (const float*)d_in[2];
  const float* Whh0 = (const float*)d_in[3];
  const float* bih0 = (const float*)d_in[4];
  const float* bhh0 = (const float*)d_in[5];
  const float* Wih1 = (const float*)d_in[6];
  const float* Whh1 = (const float*)d_in[7];
  const float* bih1 = (const float*)d_in[8];
  const float* bhh1 = (const float*)d_in[9];
  const float* Wih2 = (const float*)d_in[10];
  const float* Whh2 = (const float*)d_in[11];
  const float* bih2 = (const float*)d_in[12];
  const float* bhh2 = (const float*)d_in[13];
  const float* fc_w = (const float*)d_in[14];
  const float* fc_b = (const float*)d_in[15];
  float* out = (float*)d_out;

  // workspace carve-up (256B aligned)
  char* wsb = (char*)d_ws;
  size_t off = 0;
  auto alloc = [&](size_t bytes) -> void* {
    void* p = wsb + off;
    off = (off + bytes + 255) & ~(size_t)255;
    return p;
  };
  _Float16* raw16   = (_Float16*)alloc((size_t)B * CTX * sizeof(_Float16));
  _Float16* wih0c16 = (_Float16*)alloc((size_t)G * CTX * sizeof(_Float16));
  _Float16* whh16_0 = (_Float16*)alloc((size_t)G * H * sizeof(_Float16));
  _Float16* whh16_1 = (_Float16*)alloc((size_t)G * H * sizeof(_Float16));
  _Float16* whh16_2 = (_Float16*)alloc((size_t)G * H * sizeof(_Float16));
  _Float16* wih16_1 = (_Float16*)alloc((size_t)G * H * sizeof(_Float16));
  _Float16* wih16_2 = (_Float16*)alloc((size_t)G * H * sizeof(_Float16));
  float*    gi0base = (float*)alloc((size_t)B * G * sizeof(float));
  float*    gi      = (float*)alloc((size_t)B * G * sizeof(float));
  float*    gh      = (float*)alloc((size_t)B * G * sizeof(float));
  float*    hst     = (float*)alloc((size_t)3 * B * H * sizeof(float));
  _Float16* hst16   = (_Float16*)alloc((size_t)3 * B * H * sizeof(_Float16));
  float*    cur     = (float*)alloc((size_t)B * 2 * sizeof(float));

  const dim3 blk(256);
  const int BH = B * H;

  // one-time conversions
  cvt_f32_f16<<<(B * CTX + 255) / 256, blk, 0, stream>>>(raw_ctx, raw16, B * CTX);
  cvt_wih0_ctx<<<(G * CTX + 255) / 256, blk, 0, stream>>>(Wih0, wih0c16);
  cvt_f32_f16<<<(G * H + 255) / 256, blk, 0, stream>>>(Whh0, whh16_0, G * H);
  cvt_f32_f16<<<(G * H + 255) / 256, blk, 0, stream>>>(Whh1, whh16_1, G * H);
  cvt_f32_f16<<<(G * H + 255) / 256, blk, 0, stream>>>(Whh2, whh16_2, G * H);
  cvt_f32_f16<<<(G * H + 255) / 256, blk, 0, stream>>>(Wih1, wih16_1, G * H);
  cvt_f32_f16<<<(G * H + 255) / 256, blk, 0, stream>>>(Wih2, wih16_2, G * H);
  init_state<<<(3 * BH + 255) / 256, blk, 0, stream>>>(hst, hst16,
                                                       obs + (size_t)(T_OBS - 1) * B * 2, cur);

  const dim3 ggrid(G / 64, B / 128);
  // gi0_base = raw_ctx @ Wih0[:, :512]^T + bih0  (constant over all steps)
  gemm_wmma<CTX><<<ggrid, blk, 0, stream>>>(raw16, wih0c16, bih0, gi0base);

  for (int t = 0; t < PRED; ++t) {
    // ---- layer 0 ----
    gemm_wmma<H><<<ggrid, blk, 0, stream>>>(hst16, whh16_0, nullptr, gh);
    gru_ew<<<BH / 256, blk, 0, stream>>>(gi0base, nullptr, gh, bhh0, Wih0, cur,
                                         hst, hst16);
    // ---- layer 1 ----
    gemm_wmma<H><<<ggrid, blk, 0, stream>>>(hst16, wih16_1, nullptr, gi);
    gemm_wmma<H><<<ggrid, blk, 0, stream>>>(hst16 + BH, whh16_1, nullptr, gh);
    gru_ew<<<BH / 256, blk, 0, stream>>>(gi, bih1, gh, bhh1, nullptr, nullptr,
                                         hst + BH, hst16 + BH);
    // ---- layer 2 ----
    gemm_wmma<H><<<ggrid, blk, 0, stream>>>(hst16 + BH, wih16_2, nullptr, gi);
    gemm_wmma<H><<<ggrid, blk, 0, stream>>>(hst16 + 2 * BH, whh16_2, nullptr, gh);
    gru_ew<<<BH / 256, blk, 0, stream>>>(gi, bih2, gh, bhh2, nullptr, nullptr,
                                         hst + 2 * BH, hst16 + 2 * BH);
    // ---- projection + feedback ----
    fc_out<<<B / 256, blk, 0, stream>>>(hst + 2 * BH, fc_w, fc_b,
                                        out + (size_t)t * B * 2, cur);
  }
}
